// QuantizedConv2d_541165879627
// MI455X (gfx1250) — compile-verified, matrix operand order checked by reading, not run on hardware
//
#include <hip/hip_runtime.h>

typedef __attribute__((ext_vector_type(16))) _Float16 v16h;
typedef __attribute__((ext_vector_type(8)))  _Float16 v8h;
typedef __attribute__((ext_vector_type(8)))  float    v8f;
typedef __attribute__((ext_vector_type(4)))  float    v4f;

#define CONV_H   56
#define CONV_W   56
#define CONV_HW  3136      // 56*56
#define CIN      256
#define COUT     256
#define NIMG     32
#define M_TOTAL  (NIMG * CONV_HW)   // 100352
#define KTOT     2304               // 256*9
#define KSTEP    32
#define NSTEPS   (KTOT / KSTEP)     // 72
#define BM       128
#define BN       256
#define NTHREADS 512                // 16 waves: 2 (M) x 8 (N)
#define LDA_PITCH 40                // halfs per A row in LDS (80B: bank-spread, 16B aligned)

// ---------------------------------------------------------------------------
// Prep: dequantize int32 weights -> f16, repack [O][I][3][3] -> wb[o][k],
// k = (kh*3+kw)*256 + i  (tap-major so each K-step of 32 has a fixed tap)
// ---------------------------------------------------------------------------
__global__ void dequant_pack_kernel(const int* __restrict__ q,
                                    const float* __restrict__ w_scale,
                                    const float* __restrict__ w_min,
                                    _Float16* __restrict__ wb) {
    int t = blockIdx.x * 256 + threadIdx.x;      // 0 .. 256*2304-1
    int o = t / KTOT;
    int r = t - o * KTOT;                        // r = kk*256 + i
    int kk = r >> 8;
    int i  = r & 255;
    float s = w_scale[0];
    float m = w_min[0];
    float wv = (float)q[o * KTOT + i * 9 + kk] * s + m;
    wb[(size_t)o * KTOT + r] = (_Float16)wv;
}

// ---------------------------------------------------------------------------
// Main: implicit-GEMM conv via v_wmma_f32_16x16x32_f16.
// Block: 512 threads (16 waves, 2x8 grid). Tile: 128(M) x 256(N) — single
// N-block so each x element's im2col gather hits L2 exactly once per M-block.
// ---------------------------------------------------------------------------
__launch_bounds__(NTHREADS)
__global__ void conv_wmma_kernel(const float* __restrict__ x,
                                 const _Float16* __restrict__ wb,
                                 const float* __restrict__ bias,
                                 float* __restrict__ out) {
    // LDS: during K-loop, first 20480B = double-buffered A tile (2 x 128 x 40 halfs)
    //      after post-loop barrier, reused as 16 x (16 x 64) f32 transpose buffers
    __shared__ __align__(16) char smem[65536];
    _Float16* sA = (_Float16*)smem;

    const int tid  = threadIdx.x;
    const int wave = tid >> 5;          // 0..15
    const int lane = tid & 31;
    const int wrow = wave >> 3;         // 0..1 : M direction
    const int wcol = wave & 7;          // 0..7 : N direction

    const int m0 = blockIdx.x * BM;     // 784 blocks, exact

    // ---- loader role: thread covers (m_local = tid&127, i_locals = lquart*8 + 0..7)
    const int lm     = tid & 127;
    const int lquart = tid >> 7;        // 0..3
    const int m  = m0 + lm;
    const int n  = m / CONV_HW;
    const int p  = m - n * CONV_HW;
    const int hy = p / CONV_W;
    const int wx = p - hy * CONV_W;
    const float* xn = x + (size_t)n * (CIN * CONV_HW);

    // ---- per-lane fragment addressing
    const int l15   = lane & 15;
    const int hiL   = lane >> 4;                  // 0/1
    const int koffA = hiL ? 8 : 0;                // halfs: lanes>=16 take K{8..15,24..31}
    const int arow_base = wrow * 64 + l15;
    const int owbase = wcol * 32;
    const _Float16* wbp0 = wb + (size_t)(owbase + l15)      * KTOT + koffA;
    const _Float16* wbp1 = wb + (size_t)(owbase + 16 + l15) * KTOT + koffA;

    v8f acc[4][2] = {};

    // ---- incremental prefetch state: tap setup once per 8 K-steps
    const float* pxp = nullptr;   // tap base ptr (incl. lquart channel offset)
    bool pvalid = false;
    int  pkk = 0, pi = 0;

    auto set_tap = [&](int kk) {
        int dh = (kk >= 6) ? 1 : ((kk >= 3) ? 0 : -1);
        int dw = kk - (dh + 1) * 3 - 1;
        int hh = hy + dh, ww = wx + dw;
        pvalid = ((unsigned)hh < (unsigned)CONV_H) && ((unsigned)ww < (unsigned)CONV_W);
        pxp = xn + (ptrdiff_t)lquart * 8 * CONV_HW + (ptrdiff_t)hh * CONV_W + ww;
    };

    auto load8 = [&](const float* xp, bool valid, v8h& r) {
        #pragma unroll
        for (int j = 0; j < 8; ++j)
            r[j] = (_Float16)(valid ? xp[(ptrdiff_t)j * CONV_HW] : 0.0f);
    };

    // prologue: stage step 0 into buffer 0
    set_tap(0);
    {
        v8h r;
        load8(pxp, pvalid, r);
        *(v8h*)(sA + lm * LDA_PITCH + lquart * 8) = r;   // ds_store_b128
    }
    pi = 1;
    __syncthreads();

    for (int s = 0; s < NSTEPS; ++s) {
        const int  buf = s & 1;
        const bool havenext = (s + 1 < NSTEPS);

        // prefetch next A micro-tile from global while this step computes
        v8h nr;
        if (havenext) {
            load8(pxp + (ptrdiff_t)pi * (KSTEP * CONV_HW), pvalid, nr);
            if (++pi == 8) { pi = 0; set_tap(++pkk); }   // once per 8 steps
        }

        // A fragments from LDS (2 x ds_load_b128 per 16x16 tile)
        v16h afrag[4];
        const _Float16* sAb = sA + buf * (BM * LDA_PITCH);
        #pragma unroll
        for (int t = 0; t < 4; ++t) {
            const _Float16* pa = sAb + (arow_base + t * 16) * LDA_PITCH + koffA;
            v8h lo = *(const v8h*)pa;
            v8h hi = *(const v8h*)(pa + 16);
            afrag[t] = __builtin_shufflevector(lo, hi,
                0, 1, 2, 3, 4, 5, 6, 7, 8, 9, 10, 11, 12, 13, 14, 15);
        }

        // B fragments direct from the L2-resident packed panel (k0 = 32*s)
        v16h bfrag[2];
        {
            const _Float16* pb = wbp0 + (size_t)s * KSTEP;
            v8h lo = *(const v8h*)pb;
            v8h hi = *(const v8h*)(pb + 16);
            bfrag[0] = __builtin_shufflevector(lo, hi,
                0, 1, 2, 3, 4, 5, 6, 7, 8, 9, 10, 11, 12, 13, 14, 15);
        }
        {
            const _Float16* pb = wbp1 + (size_t)s * KSTEP;
            v8h lo = *(const v8h*)pb;
            v8h hi = *(const v8h*)(pb + 16);
            bfrag[1] = __builtin_shufflevector(lo, hi,
                0, 1, 2, 3, 4, 5, 6, 7, 8, 9, 10, 11, 12, 13, 14, 15);
        }

        // 8 WMMAs: wave tile 64(M) x 32(N)
        #pragma unroll
        for (int t = 0; t < 4; ++t)
            #pragma unroll
            for (int u = 0; u < 2; ++u)
                acc[t][u] = __builtin_amdgcn_wmma_f32_16x16x32_f16(
                    false, afrag[t], false, bfrag[u],
                    (short)0, acc[t][u], false, false);

        if (havenext) {
            *(v8h*)(sA + (buf ^ 1) * (BM * LDA_PITCH) + lm * LDA_PITCH + lquart * 8) = nr;
            __syncthreads();
        }
    }

    // ---- epilogue: bias + per-wave LDS transpose (two 16x64 passes, 4KB/wave)
    //      so global stores are 16B-vector, contiguous along W.
    __syncthreads();                              // sA region no longer needed
    float* sCw = (float*)smem + wave * 1024;      // 16(o) x 64(m) per wave
    const int mwbase = m0 + wrow * 64;

    #pragma unroll
    for (int u = 0; u < 2; ++u) {
        const float bv = bias[owbase + u * 16 + l15];
        #pragma unroll
        for (int t = 0; t < 4; ++t) {
            const int mlb = t * 16 + hiL * 8;
            #pragma unroll
            for (int v = 0; v < 8; ++v)
                sCw[l15 * 64 + mlb + v] = acc[t][u][v] + bv;
        }
        // in-wave DS ordering makes the transpose visible without a barrier
        #pragma unroll
        for (int idx = 0; idx < 8; ++idx) {
            int g  = idx * 32 + lane;             // 256 groups of 4 floats per pass
            int ol = g >> 4;
            int ml = (g & 15) * 4;
            v4f val = *(const v4f*)&sCw[ol * 64 + ml];
            int mm = mwbase + ml;
            int nn = mm / CONV_HW;
            int pp = mm - nn * CONV_HW;           // 4-aligned; group never crosses image
            float* op = out + (size_t)nn * (COUT * CONV_HW)
                            + (size_t)(owbase + u * 16 + ol) * CONV_HW + pp;
            *(v4f*)op = val;                      // global_store_b128 along W
        }
    }
}

extern "C" void kernel_launch(void* const* d_in, const int* in_sizes, int n_in,
                              void* d_out, int out_size, void* d_ws, size_t ws_size,
                              hipStream_t stream) {
    const float* x       = (const float*)d_in[0];
    const int*   qweight = (const int*)d_in[1];
    const float* w_scale = (const float*)d_in[2];
    const float* w_min   = (const float*)d_in[3];
    const float* bias    = (const float*)d_in[4];
    float*       out     = (float*)d_out;

    _Float16* wb = (_Float16*)d_ws;    // 256*2304 f16 = 1.13 MB packed weight panel

    // 256*2304 = 589824 weight elements
    dequant_pack_kernel<<<2304, 256, 0, stream>>>(qweight, w_scale, w_min, wb);

    dim3 grid(M_TOTAL / BM);           // 784 blocks, 512 threads each
    conv_wmma_kernel<<<grid, NTHREADS, 0, stream>>>(x, wb, bias, out);
}
